// TestModel_18588618457382
// MI455X (gfx1250) — matrix operand order repaired and checked by reading, not compiled, not run on hardware
//
#include <hip/hip_runtime.h>
#include <hip/hip_bf16.h>

// ---------------------------------------------------------------------------
// CDNA5 (gfx1250) recursive structure-VAE forward pass.
//  * batched matmuls -> v_wmma_f32_16x16x32_f16 (f32 accumulate), all
//    full-tile (M padded to multiples of 16 -> no exec-mask predication)
//  * sequential tree chains -> single-workgroup matvec loops; 256KB of f16
//    chain weights DMA'd into LDS via the Tensor Data Mover (TDM) and
//    synchronized with s_wait_tensorcnt (MI455X: 320KB LDS per WG)
// ---------------------------------------------------------------------------

typedef __attribute__((ext_vector_type(16))) _Float16 v16h;
typedef __attribute__((ext_vector_type(8)))  _Float16 v8h;
typedef __attribute__((ext_vector_type(8)))  float    v8f;
typedef __attribute__((ext_vector_type(4)))  unsigned int tdm_v4u;
typedef __attribute__((ext_vector_type(8)))  int          tdm_v8i;
typedef __attribute__((ext_vector_type(4)))  int          tdm_v4i;

#define LATD 128
#define HIDD 256
#define SYMD 32
#define BOXD 64
#define CATD 3

#if __has_include(<hip/amd_detail/amd_gfx1250_TDM.h>)
#define VAE_TDM_6ARG 1
#endif

// ---------------------------------------------------------------------------
// TDM: 1-D DMA of `bytes` (multiple of 8) from global f16 image -> LDS.
// D# per CDNA5 ISA ch.8: group0 = {count/type/addr}, group1 = dims/strides.
// data_size = 8B elements; tile = tensor = one row of n8 elements.
// ---------------------------------------------------------------------------
__device__ __forceinline__ void vae_tdm_load_1d(const _Float16* __restrict__ gsrc,
                                                unsigned ldsByteAddr, unsigned bytes)
{
    unsigned long long ga = (unsigned long long)(size_t)gsrc;
    unsigned n8 = bytes >> 3;
    tdm_v4u g0 = {};
    g0[0] = 1u;                                         // count=1 (user D#)
    g0[1] = ldsByteAddr;                                // lds_addr
    g0[2] = (unsigned)(ga & 0xffffffffu);               // global_addr[31:0]
    g0[3] = (unsigned)((ga >> 32) & 0x01ffffffu)        // global_addr[56:32]
          | (2u << 30);                                 // type=2 ("image")
    tdm_v8i g1 = {};
    g1[0] = (int)(3u << 16);                            // data_size = 8 bytes
    g1[1] = (int)((n8 & 0xffffu) << 16);                // tensor_dim0[15:0]
    g1[2] = (int)(((n8 >> 16) & 0xffffu) | (1u << 16)); // dim0[31:16] | tensor_dim1=1
    g1[3] = (int)((n8 & 0xffffu) << 16);                // tile_dim0
    g1[4] = 0;                                          // tile_dim1/2 unused
    g1[5] = (int)n8;                                    // tensor_dim0_stride[31:0]
    g1[6] = 0; g1[7] = 0;
    tdm_v4i z4 = {};
#ifdef VAE_TDM_6ARG
    tdm_v8i z8 = {};
    __builtin_amdgcn_tensor_load_to_lds(g0, g1, z4, z4, z8, 0);
#else
    __builtin_amdgcn_tensor_load_to_lds(g0, g1, z4, z4, 0);
#endif
}

// ---------------------------------------------------------------------------
// WMMA fragment loaders (layouts per CDNA5 ISA 7.12.2)
// ---------------------------------------------------------------------------
__device__ __forceinline__ v16h vae_ldA(const _Float16* __restrict__ A,
                                        int K, int rowA, int kb, int khalf)
{
    const _Float16* ap = A + (size_t)rowA * K + kb + khalf * 8;
    v8h a0 = *(const v8h*)ap;
    v8h a1 = *(const v8h*)(ap + 16);
    v16h a;
    #pragma unroll
    for (int i = 0; i < 8; ++i) { a[i] = a0[i]; a[8 + i] = a1[i]; }
    return a;
}
__device__ __forceinline__ v16h vae_ldB(const _Float16* __restrict__ Bt,
                                        int K, int colB, int kb, int khalf)
{
    const _Float16* bp = Bt + (size_t)colB * K + kb + khalf * 8;
    v8h b0 = *(const v8h*)bp;
    v8h b1 = *(const v8h*)(bp + 16);
    v16h b;
    #pragma unroll
    for (int i = 0; i < 8; ++i) { b[i] = b0[i]; b[8 + i] = b1[i]; }
    return b;
}

// ---------------------------------------------------------------------------
// WMMA GEMM:  C = [tanh]( A(f16, MxK) @ Bt^T(f16, Npad x K) + bias )
// M is always a multiple of 16 (callers pad) -> no predication anywhere.
// Dual accumulators break WMMA->WMMA RAW hazard chains.
// ---------------------------------------------------------------------------
__global__ __launch_bounds__(256) void vae_gemm_wmma(
    const _Float16* __restrict__ A, const _Float16* __restrict__ Bt,
    const float* __restrict__ bias,
    float* __restrict__ C32, _Float16* __restrict__ C16,
    int M, int K, int Npad, int Nreal, int doTanh)
{
    const int lane   = threadIdx.x & 31;
    const int wave   = threadIdx.x >> 5;
    const int Ntiles = Npad >> 4;
    const int Mtiles = M >> 4;
    const int tile   = blockIdx.x * 8 + wave;
    if (tile >= Mtiles * Ntiles) return;
    const int tm = tile / Ntiles;
    const int tn = tile % Ntiles;

    const int l16   = lane & 15;
    const int khalf = lane >> 4;
    const int rowA  = tm * 16 + l16;
    const int colB  = tn * 16 + l16;

    v8f acc0 = {};
    v8f acc1 = {};
    for (int kb = 0; kb < K; kb += 64) {
        v16h a0 = vae_ldA(A, K, rowA, kb, khalf);
        v16h b0 = vae_ldB(Bt, K, colB, kb, khalf);
        acc0 = __builtin_amdgcn_wmma_f32_16x16x32_f16(
            false, a0, false, b0, (short)0, acc0, false, false);
        if (kb + 32 < K) {
            v16h a1 = vae_ldA(A, K, rowA, kb + 32, khalf);
            v16h b1 = vae_ldB(Bt, K, colB, kb + 32, khalf);
            acc1 = __builtin_amdgcn_wmma_f32_16x16x32_f16(
                false, a1, false, b1, (short)0, acc1, false, false);
        }
    }

    const int col = tn * 16 + l16;
    const float bv = (bias && col < Nreal) ? bias[col] : 0.0f;
    #pragma unroll
    for (int v = 0; v < 8; ++v) {
        const int row = tm * 16 + (khalf << 3) + v;
        float val = acc0[v] + acc1[v] + bv;
        if (doTanh) val = tanhf(val);
        if (C32) C32[(size_t)row * Npad + col] = val;
        if (C16) C16[(size_t)row * Npad + col] = (_Float16)val;
    }
}

// ---------------------------------------------------------------------------
// Conversion / transpose / image-building kernels
// ---------------------------------------------------------------------------
__global__ void vae_cvt_transpose(const float* __restrict__ src, int srcStride,
                                  int rowOff, int K, int Nreal, int Npad,
                                  _Float16* __restrict__ dst)
{
    int idx = blockIdx.x * blockDim.x + threadIdx.x;
    if (idx >= Npad * K) return;
    int n = idx / K, k = idx % K;
    dst[idx] = (n < Nreal) ? (_Float16)src[(size_t)(rowOff + k) * srcStride + n]
                           : (_Float16)0.0f;
}

__global__ void vae_cvt_shapeT(const float* __restrict__ shape,
                               _Float16* __restrict__ aLeaf, int leafN)
{
    int idx = blockIdx.x * blockDim.x + threadIdx.x;
    if (idx >= leafN * BOXD) return;
    int i = idx >> 6, b = idx & 63;
    aLeaf[idx] = (_Float16)shape[(size_t)b * leafN + i];
}

__global__ void vae_cvt_f16_pad(const float* __restrict__ src,
                                _Float16* __restrict__ dst, int nReal, int nTot)
{
    int idx = blockIdx.x * blockDim.x + threadIdx.x;
    if (idx >= nTot) return;
    dst[idx] = (idx < nReal) ? (_Float16)src[idx] : (_Float16)0.0f;
}

__global__ void vae_zero_f32(float* __restrict__ p, int n)
{
    int idx = blockIdx.x * blockDim.x + threadIdx.x;
    if (idx < n) p[idx] = 0.0f;
}

// f16 LDS images for the chain kernels (exact LDS layout, built once)
__global__ void vae_build_encimg(const float* __restrict__ Wa1,
                                 const float* __restrict__ Wa2,
                                 const float* __restrict__ Ws1,
                                 const float* __restrict__ Ws2,
                                 _Float16* __restrict__ img)
{
    int i = blockIdx.x * blockDim.x + threadIdx.x;
    if (i >= 32768) return;
    img[i]         = (_Float16)Wa1[i];   // Wa1 rows 0..127 (contiguous)
    img[32768 + i] = (_Float16)Wa2[i];
    img[65536 + i] = (_Float16)Ws1[i];   // Ws1 rows 0..127
    img[98304 + i] = (_Float16)Ws2[i];
}

__global__ void vae_build_decimg(const float* __restrict__ Da1,
                                 const float* __restrict__ Da2,
                                 const float* __restrict__ Ds1,
                                 const float* __restrict__ Ds2,
                                 _Float16* __restrict__ img)
{
    int i = blockIdx.x * blockDim.x + threadIdx.x;
    if (i >= 32768) return;
    const int k = i >> 7, c = i & 127;
    img[i]         = (_Float16)Da1[i];
    img[32768 + i] = (_Float16)Da2[(size_t)k * (2 * LATD + CATD) + c];
    img[65536 + i] = (_Float16)Ds1[i];
    img[98304 + i] = (_Float16)Ds2[(size_t)k * (LATD + SYMD + CATD) + c];
}

// ---------------------------------------------------------------------------
// Matvec dot helpers
// ---------------------------------------------------------------------------
__device__ __forceinline__ float vae_dot(const float* __restrict__ xs, int n,
                                         const float* __restrict__ W,
                                         int stride, int c)
{
    float a0 = 0.f, a1 = 0.f, a2 = 0.f, a3 = 0.f;
    #pragma unroll 8
    for (int k = 0; k < n; k += 4) {
        a0 = fmaf(xs[k + 0], W[(size_t)(k + 0) * stride + c], a0);
        a1 = fmaf(xs[k + 1], W[(size_t)(k + 1) * stride + c], a1);
        a2 = fmaf(xs[k + 2], W[(size_t)(k + 2) * stride + c], a2);
        a3 = fmaf(xs[k + 3], W[(size_t)(k + 3) * stride + c], a3);
    }
    return (a0 + a1) + (a2 + a3);
}

__device__ __forceinline__ float vae_dot_h(const float* __restrict__ xs, int n,
                                           const _Float16* __restrict__ W,
                                           int stride, int c)
{
    float a0 = 0.f, a1 = 0.f, a2 = 0.f, a3 = 0.f;
    #pragma unroll 8
    for (int k = 0; k < n; k += 4) {
        a0 = fmaf(xs[k + 0], (float)W[(k + 0) * stride + c], a0);
        a1 = fmaf(xs[k + 1], (float)W[(k + 1) * stride + c], a1);
        a2 = fmaf(xs[k + 2], (float)W[(k + 2) * stride + c], a2);
        a3 = fmaf(xs[k + 3], (float)W[(k + 3) * stride + c], a3);
    }
    return (a0 + a1) + (a2 + a3);
}

// ---------------------------------------------------------------------------
// Encoder chain (single workgroup). 256KB f16 weights TDM-loaded into LDS:
//   [0) Wa1_top  [32K) Wa2  [64K) Ws1_top  [96K) Ws2   (element offsets)
// Step: t = tanh(x @ W1_top + pre[j]);  x = tanh(t @ W2 + b2)
// ---------------------------------------------------------------------------
__global__ __launch_bounds__(256) void vae_enc_chain(
    const int* __restrict__ kids, const float* __restrict__ hEncLeaf,
    const float* __restrict__ leafPre, const float* __restrict__ paramPre,
    const _Float16* __restrict__ wImg,
    const float* __restrict__ bWa2, const float* __restrict__ bWs2,
    float* __restrict__ rootH, int N, int leafN)
{
    extern __shared__ _Float16 wEnc[];         // 131072 halves = 256KB dynamic
    __shared__ float x[LATD];
    __shared__ float t[HIDD];
    __shared__ float b2a[LATD];
    __shared__ float b2s[LATD];
    const int tid = threadIdx.x;

    if (tid == 0)
        vae_tdm_load_1d(wImg, __builtin_amdgcn_groupstaticsize(), 131072u * 2u);
    if (tid < LATD) { b2a[tid] = bWa2[tid]; b2s[tid] = bWs2[tid]; }
    int ia0 = kids[leafN * 3 + 0] - 1; if (ia0 < 0) ia0 = 0;
    if (tid < LATD) x[tid] = hEncLeaf[(size_t)ia0 * LATD + tid];
    __builtin_amdgcn_s_wait_tensorcnt(0);
    __syncthreads();

    for (int j = leafN; j < N; ++j) {
        const int k0 = kids[j * 3 + 0];
        const int k1 = kids[j * 3 + 1];
        const int k2 = kids[j * 3 + 2];
        const bool sym = (k2 == 1);
        const _Float16* W1 = sym ? (wEnc + 65536) : wEnc;
        const _Float16* W2 = sym ? (wEnc + 98304) : (wEnc + 32768);
        const float*    b2 = sym ? b2s : b2a;
        const float*   pre = sym ? (paramPre + (size_t)(k0 - 1 - leafN) * HIDD)
                                 : (leafPre  + (size_t)(k1 - 1) * HIDD);
        // prefetch next node's pre row while this step computes
        if (j + 1 < N) {
            const int nk0 = kids[(j + 1) * 3 + 0];
            const int nk1 = kids[(j + 1) * 3 + 1];
            const int nk2 = kids[(j + 1) * 3 + 2];
            const float* npre = (nk2 == 1)
                ? (paramPre + (size_t)(nk0 - 1 - leafN) * HIDD)
                : (leafPre  + (size_t)(nk1 - 1) * HIDD);
            __builtin_prefetch(npre + tid, 0, 0);
        }
        t[tid] = tanhf(vae_dot_h(x, LATD, W1, HIDD, tid) + pre[tid]);
        __syncthreads();
        if (tid < LATD)
            x[tid] = tanhf(vae_dot_h(t, HIDD, W2, LATD, tid) + b2[tid]);
        __syncthreads();
    }
    if (tid < LATD) rootH[tid] = x[tid];
}

// ---------------------------------------------------------------------------
// Root head: re1/re2 -> reparameterize -> rd2/rd1 -> hmid[N-1]  (one shot)
// ---------------------------------------------------------------------------
__global__ __launch_bounds__(256) void vae_root_head(
    const float* __restrict__ rootH, const float* __restrict__ eps,
    const float* __restrict__ We1, const float* __restrict__ bWe1,
    const float* __restrict__ We2, const float* __restrict__ bWe2,
    const float* __restrict__ Wd2, const float* __restrict__ bWd2,
    const float* __restrict__ Wd1, const float* __restrict__ bWd1,
    float* __restrict__ hmid, int N)
{
    __shared__ float x[LATD];
    __shared__ float t[HIDD];
    __shared__ float u[HIDD];
    const int tid = threadIdx.x;

    if (tid < LATD) x[tid] = rootH[tid];
    __syncthreads();
    t[tid] = tanhf(vae_dot(x, LATD, We1, HIDD, tid) + bWe1[tid]);   // re1
    __syncthreads();
    u[tid] = vae_dot(t, HIDD, We2, HIDD, tid) + bWe2[tid];          // re2
    __syncthreads();
    if (tid < LATD)                                                  // z
        x[tid] = u[tid] + expf(u[LATD + tid] * 0.5f) * eps[tid];
    __syncthreads();
    t[tid] = tanhf(vae_dot(x, LATD, Wd2, HIDD, tid) + bWd2[tid]);   // rd2
    __syncthreads();
    if (tid < LATD)                                                  // rd1
        hmid[(size_t)(N - 1) * LATD + tid] =
            tanhf(vae_dot(t, HIDD, Wd1, LATD, tid) + bWd1[tid]);
}

// ---------------------------------------------------------------------------
// Decoder chain (single workgroup). LDS (TDM-loaded, element offsets):
//   [0) Da1  [32K) Da2 cols 0..127  [64K) Ds1  [96K) Ds2 cols 0..127
// ---------------------------------------------------------------------------
__global__ __launch_bounds__(256) void vae_dec_chain(
    const int* __restrict__ kids, float* __restrict__ hmid,
    const _Float16* __restrict__ wImg,
    const float* __restrict__ bDa1, const float* __restrict__ bDa2,
    const float* __restrict__ bDs1, const float* __restrict__ bDs2,
    int N, int leafN)
{
    extern __shared__ _Float16 wDec[];         // 131072 halves = 256KB dynamic
    __shared__ float x[LATD];
    __shared__ float t[HIDD];
    __shared__ float b1a[HIDD];
    __shared__ float b1s[HIDD];
    __shared__ float b2a[LATD];
    __shared__ float b2s[LATD];
    const int tid = threadIdx.x;

    if (tid == 0)
        vae_tdm_load_1d(wImg, __builtin_amdgcn_groupstaticsize(), 131072u * 2u);
    b1a[tid] = bDa1[tid];
    b1s[tid] = bDs1[tid];
    if (tid < LATD) { b2a[tid] = bDa2[tid]; b2s[tid] = bDs2[tid]; }
    if (tid < LATD) x[tid] = hmid[(size_t)(N - 1) * LATD + tid];
    __builtin_amdgcn_s_wait_tensorcnt(0);
    __syncthreads();

    for (int j = N - 1; j >= leafN; --j) {
        const int k0 = kids[j * 3 + 0];
        const int k2 = kids[j * 3 + 2];
        const bool sym = (k2 == 1);
        const _Float16* D1 = sym ? (wDec + 65536) : wDec;
        const _Float16* D2 = sym ? (wDec + 98304) : (wDec + 32768);
        const float* b1 = sym ? b1s : b1a;
        const float* b2 = sym ? b2s : b2a;

        t[tid] = tanhf(vae_dot_h(x, LATD, D1, HIDD, tid) + b1[tid]);
        __syncthreads();
        if (tid < LATD) {
            float hn = tanhf(vae_dot_h(t, HIDD, D2, LATD, tid) + b2[tid]);
            int ia = k0 - 1; if (ia < 0) ia = 0;
            hmid[(size_t)ia * LATD + tid] = hn;
            x[tid] = hn;
        }
        __syncthreads();
    }
}

// ---------------------------------------------------------------------------
// Decoder post-pass for internal nodes (parallel across ~10921 blocks)
// ---------------------------------------------------------------------------
__global__ __launch_bounds__(256) void vae_dec_post(
    const int* __restrict__ kids, float* __restrict__ hmid,
    const float* __restrict__ Da1, const float* __restrict__ bDa1,
    const float* __restrict__ Da2, const float* __restrict__ bDa2,
    const float* __restrict__ Ds1, const float* __restrict__ bDs1,
    const float* __restrict__ Ds2, const float* __restrict__ bDs2,
    float* __restrict__ pOut, float* __restrict__ nclr, int N, int leafN)
{
    const int j = leafN + blockIdx.x;
    __shared__ float x[LATD];
    __shared__ float t[HIDD];
    const int tid = threadIdx.x;

    const int k0 = kids[j * 3 + 0];
    const int k1 = kids[j * 3 + 1];
    const int k2 = kids[j * 3 + 2];
    const bool sym = (k2 == 1);
    const float* D1 = sym ? Ds1 : Da1;
    const float* b1 = sym ? bDs1 : bDa1;
    const float* D2 = sym ? Ds2 : Da2;
    const float* b2 = sym ? bDs2 : bDa2;
    const int nc2   = sym ? (LATD + SYMD + CATD) : (2 * LATD + CATD);

    if (tid < LATD) x[tid] = hmid[(size_t)j * LATD + tid];
    __syncthreads();
    t[tid] = tanhf(vae_dot(x, LATD, D1, HIDD, tid) + b1[tid]);
    __syncthreads();

    const int extra = nc2 - LATD;              // adj: 131, sym: 35
    if (tid < extra) {
        const int c = LATD + tid;
        float val = tanhf(vae_dot(t, HIDD, D2, nc2, c) + b2[c]);
        if (sym) {
            if (tid < SYMD) {
                int ia = k0 - 1; if (ia < 0) ia = 0;
                pOut[(size_t)ia * SYMD + tid] = val;
            } else {
                nclr[(size_t)j * CATD + (tid - SYMD)] = val;
            }
        } else {
            if (tid < LATD) {
                int ib = k1 - 1; if (ib < 0) ib = 0;
                hmid[(size_t)ib * LATD + tid] = val;   // leaf child latent
            } else {
                nclr[(size_t)j * CATD + (tid - LATD)] = val;
            }
        }
    }
}

// ---------------------------------------------------------------------------
// Leaf output scatter + tail
// ---------------------------------------------------------------------------
__global__ void vae_scatter_leaf(const float* __restrict__ outLeaf,
                                 float* __restrict__ hBox,
                                 float* __restrict__ nclr, int leafN)
{
    int idx = blockIdx.x * blockDim.x + threadIdx.x;
    if (idx >= leafN * (BOXD + CATD)) return;
    int l = idx / (BOXD + CATD), c = idx % (BOXD + CATD);
    float v = outLeaf[(size_t)l * 80 + c];
    if (c < BOXD) hBox[(size_t)l * BOXD + c] = v;
    else          nclr[(size_t)l * CATD + (c - BOXD)] = v;
}

__global__ void vae_tail(const int* __restrict__ kids,
                         const float* __restrict__ param,
                         float* __restrict__ paramGT, int* __restrict__ gt,
                         int N)
{
    int i = blockIdx.x * blockDim.x + threadIdx.x;
    if (i >= N) return;
    const int k0 = kids[i * 3 + 0];
    const int k2 = kids[i * 3 + 2];
    const bool leaf = (k0 == 0);
    const bool sym  = (!leaf) && (k2 == 1);
    int ia = k0 - 1; if (ia < 0) ia = 0; if (ia > N - 1) ia = N - 1;
    for (int s = 0; s < SYMD; ++s)
        paramGT[(size_t)i * SYMD + s] = sym ? param[(size_t)ia * SYMD + s] : 0.0f;
    gt[i] = leaf ? 0 : (sym ? 2 : 1);
}

// ---------------------------------------------------------------------------
// Host-side orchestration
// ---------------------------------------------------------------------------
extern "C" void kernel_launch(void* const* d_in, const int* in_sizes, int n_in,
                              void* d_out, int out_size, void* d_ws, size_t ws_size,
                              hipStream_t stream)
{
    const int*   kids  = (const int*)d_in[0];
    const float* shape = (const float*)d_in[1];
    const float* param = (const float*)d_in[2];
    const float* eps   = (const float*)d_in[3];
    const float* Da1 = (const float*)d_in[4];   const float* bDa1 = (const float*)d_in[5];
    const float* Da2 = (const float*)d_in[6];   const float* bDa2 = (const float*)d_in[7];
    const float* Dl1 = (const float*)d_in[8];   const float* bDl1 = (const float*)d_in[9];
    const float* Dl2 = (const float*)d_in[10];  const float* bDl2 = (const float*)d_in[11];
    const float* Ds1 = (const float*)d_in[12];  const float* bDs1 = (const float*)d_in[13];
    const float* Ds2 = (const float*)d_in[14];  const float* bDs2 = (const float*)d_in[15];
    const float* Wa1 = (const float*)d_in[16];  const float* bWa1 = (const float*)d_in[17];
    const float* Wa2 = (const float*)d_in[18];  const float* bWa2 = (const float*)d_in[19];
    const float* Wd1 = (const float*)d_in[20];  const float* bWd1 = (const float*)d_in[21];
    const float* Wd2 = (const float*)d_in[22];  const float* bWd2 = (const float*)d_in[23];
    const float* We1 = (const float*)d_in[24];  const float* bWe1 = (const float*)d_in[25];
    const float* We2 = (const float*)d_in[26];  const float* bWe2 = (const float*)d_in[27];
    const float* Wl  = (const float*)d_in[28];  const float* bWl  = (const float*)d_in[29];
    const float* Ws1 = (const float*)d_in[30];  const float* bWs1 = (const float*)d_in[31];
    const float* Ws2 = (const float*)d_in[32];  const float* bWs2 = (const float*)d_in[33];

    const int N     = in_sizes[0] / 3;
    const int leafN = in_sizes[1] / BOXD;
    const int NI    = N - leafN;
    const int NIpad = (NI + 15) & ~15;

    // Output layout (flat, in reference return order)
    float* out     = (float*)d_out;
    float* hBox    = out;                                // leafN*64
    float* pOut    = hBox + (size_t)leafN * BOXD;        // N*32
    float* paramGT = pOut + (size_t)N * SYMD;            // N*32
    float* nclr    = paramGT + (size_t)N * SYMD;         // N*3
    int*   gt      = (int*)(nclr + (size_t)N * CATD);    // N

    // Workspace bump allocator
    char* ws = (char*)d_ws;
    size_t off = 0;
    auto alloc = [&](size_t bytes) -> void* {
        void* p = ws + off;
        off = (off + bytes + 255) & ~(size_t)255;
        return p;
    };
    _Float16* aLeaf    = (_Float16*)alloc((size_t)leafN * BOXD * 2);
    _Float16* WlT      = (_Float16*)alloc((size_t)LATD * BOXD * 2);
    float*    hEnc     = (float*)   alloc((size_t)leafN * LATD * 4);
    _Float16* hEnc16   = (_Float16*)alloc((size_t)leafN * LATD * 2);
    _Float16* Wa1botT  = (_Float16*)alloc((size_t)HIDD * LATD * 2);
    _Float16* Ws1botT  = (_Float16*)alloc((size_t)HIDD * SYMD * 2);
    _Float16* param16  = (_Float16*)alloc((size_t)NIpad * SYMD * 2);
    float*    leafPre  = (float*)   alloc((size_t)leafN * HIDD * 4);
    float*    paramPre = (float*)   alloc((size_t)NIpad * HIDD * 4);
    float*    rootH    = (float*)   alloc((size_t)LATD * 4);
    float*    hmid     = (float*)   alloc((size_t)N * LATD * 4);
    _Float16* hmid16   = (_Float16*)alloc((size_t)leafN * LATD * 2);
    _Float16* Dl1T     = (_Float16*)alloc((size_t)HIDD * LATD * 2);
    _Float16* t16      = (_Float16*)alloc((size_t)leafN * HIDD * 2);
    _Float16* Dl2T     = (_Float16*)alloc((size_t)80 * HIDD * 2);
    float*    outLeaf  = (float*)   alloc((size_t)leafN * 80 * 4);
    _Float16* encImg   = (_Float16*)alloc((size_t)131072 * 2);
    _Float16* decImg   = (_Float16*)alloc((size_t)131072 * 2);

    const int T = 256;
    const int chainLds = 131072 * 2;   // 256KB dynamic LDS for the chains
    (void)hipFuncSetAttribute(reinterpret_cast<const void*>(vae_enc_chain),
                              hipFuncAttributeMaxDynamicSharedMemorySize, chainLds);
    (void)hipFuncSetAttribute(reinterpret_cast<const void*>(vae_dec_chain),
                              hipFuncAttributeMaxDynamicSharedMemorySize, chainLds);

    auto grid1 = [&](int n) { return dim3((n + T - 1) / T); };
    auto gemm = [&](const _Float16* A, const _Float16* Bt, const float* bias,
                    float* C32, _Float16* C16, int M, int K, int Npad, int Nreal,
                    int doTanh) {
        int tiles  = (M / 16) * (Npad / 16);
        int blocks = (tiles + 7) / 8;
        vae_gemm_wmma<<<blocks, 256, 0, stream>>>(A, Bt, bias, C32, C16,
                                                  M, K, Npad, Nreal, doTanh);
    };

    // --- phase 0: zero pOut (only partially written later) ------------------
    vae_zero_f32<<<grid1(N * SYMD), T, 0, stream>>>(pOut, N * SYMD);

    // --- phase 1: conversions / transposes / LDS images ---------------------
    vae_cvt_shapeT<<<grid1(leafN * BOXD), T, 0, stream>>>(shape, aLeaf, leafN);
    vae_cvt_f16_pad<<<grid1(NIpad * SYMD), T, 0, stream>>>(
        param + (size_t)leafN * SYMD, param16, NI * SYMD, NIpad * SYMD);
    vae_cvt_transpose<<<grid1(LATD * BOXD), T, 0, stream>>>(Wl, LATD, 0, BOXD, LATD, LATD, WlT);
    vae_cvt_transpose<<<grid1(HIDD * LATD), T, 0, stream>>>(Wa1, HIDD, LATD, LATD, HIDD, HIDD, Wa1botT);
    vae_cvt_transpose<<<grid1(HIDD * SYMD), T, 0, stream>>>(Ws1, HIDD, LATD, SYMD, HIDD, HIDD, Ws1botT);
    vae_cvt_transpose<<<grid1(HIDD * LATD), T, 0, stream>>>(Dl1, HIDD, 0, LATD, HIDD, HIDD, Dl1T);
    vae_cvt_transpose<<<grid1(80 * HIDD), T, 0, stream>>>(Dl2, BOXD + CATD, 0, HIDD, BOXD + CATD, 80, Dl2T);
    vae_build_encimg<<<grid1(32768), T, 0, stream>>>(Wa1, Wa2, Ws1, Ws2, encImg);
    vae_build_decimg<<<grid1(32768), T, 0, stream>>>(Da1, Da2, Ds1, Ds2, decImg);

    // --- phase 2: batched leaf encode + chain precomputation (WMMA) ---------
    gemm(aLeaf, WlT, bWl, hEnc, hEnc16, leafN, BOXD, LATD, LATD, 1);
    gemm(hEnc16, Wa1botT, bWa1, leafPre, nullptr, leafN, LATD, HIDD, HIDD, 0);
    gemm(param16, Ws1botT, bWs1, paramPre, nullptr, NIpad, SYMD, HIDD, HIDD, 0);

    // --- phase 3: sequential encoder chain (TDM-staged LDS weights) ---------
    vae_enc_chain<<<1, 256, chainLds, stream>>>(kids, hEnc, leafPre, paramPre,
                                                encImg, bWa2, bWs2,
                                                rootH, N, leafN);

    // --- phase 4: VAE root head ---------------------------------------------
    vae_root_head<<<1, 256, 0, stream>>>(rootH, eps, We1, bWe1, We2, bWe2,
                                         Wd2, bWd2, Wd1, bWd1, hmid, N);

    // --- phase 5: sequential decoder chain (TDM-staged LDS weights) ---------
    vae_dec_chain<<<1, 256, chainLds, stream>>>(kids, hmid, decImg,
                                                bDa1, bDa2, bDs1, bDs2,
                                                N, leafN);

    // --- phase 6: parallel decoder post-pass (wide outputs) ------------------
    vae_dec_post<<<NI, 256, 0, stream>>>(kids, hmid, Da1, bDa1, Da2, bDa2,
                                         Ds1, bDs1, Ds2, bDs2, pOut, nclr,
                                         N, leafN);

    // --- phase 7: batched leaf decode (WMMA) ---------------------------------
    vae_cvt_f16_pad<<<grid1(leafN * LATD), T, 0, stream>>>(hmid, hmid16,
                                                           leafN * LATD, leafN * LATD);
    gemm(hmid16, Dl1T, bDl1, nullptr, t16, leafN, LATD, HIDD, HIDD, 1);
    gemm(t16, Dl2T, bDl2, outLeaf, nullptr, leafN, HIDD, 80, BOXD + CATD, 1);
    vae_scatter_leaf<<<grid1(leafN * (BOXD + CATD)), T, 0, stream>>>(outLeaf, hBox, nclr, leafN);

    // --- phase 8: paramGT + gt -----------------------------------------------
    vae_tail<<<grid1(N), T, 0, stream>>>(kids, param, paramGT, gt, N);

    (void)n_in; (void)out_size; (void)ws_size;
}
